// KernelNN_1623497638451
// MI455X (gfx1250) — matrix operand order repaired and checked by reading, not compiled, not run on hardware
//
#include <hip/hip_runtime.h>
#include <hip/hip_bf16.h>

// ---------------------------------------------------------------------------
// CDNA5 / gfx1250 implementation of the NNConv graph network.
//   - Kernel-MLP GEMMs run on v_wmma_f32_16x16x32_f16 (wave32 WMMA),
//     double-buffered LDS, A-tile staged with global_load_async_to_lds_b128
//     (ASYNCcnt) when available, B-tile software-pipelined through registers.
//   - Per-edge [64x64] weights stored f16, permuted [E,o,d] to halve HBM
//     traffic of the 6 message-passing sweeps (the true bottleneck).
//   - Scatter-mean via global_atomic_add_f32.
// ---------------------------------------------------------------------------

typedef __attribute__((ext_vector_type(16))) _Float16 v16h;
typedef __attribute__((ext_vector_type(8)))  float    v8f;
typedef __attribute__((ext_vector_type(4)))  int      v4i;

#define WIDTH 64
#define KW    1024

#if defined(__AMDGCN__) && __has_builtin(__builtin_amdgcn_global_load_async_to_lds_b128)
#define HAVE_ASYNC_LDS 1
#else
#define HAVE_ASYNC_LDS 0
#endif

#if HAVE_ASYNC_LDS
typedef __attribute__((address_space(1))) v4i* gv4i_p;   // global v4i*
typedef __attribute__((address_space(3))) v4i* lv4i_p;   // LDS v4i*

__device__ __forceinline__ void async_cp16(void* lds, const void* gp) {
    // Generic->AS casts via integers: global is identity-mapped; the low 32
    // bits of a generic LDS pointer are the AS3 offset (flat aperture rule).
    __builtin_amdgcn_global_load_async_to_lds_b128(
        (gv4i_p)(unsigned long long)(size_t)gp,
        (lv4i_p)(unsigned int)(size_t)lds,
        0, 0);
}
__device__ __forceinline__ void wait_async0() {
#if __has_builtin(__builtin_amdgcn_s_wait_asynccnt)
    __builtin_amdgcn_s_wait_asynccnt(0);
#else
    asm volatile("s_wait_asynccnt 0x0" ::: "memory");
#endif
}
#endif

// ---------------------------------------------------------------------------
// f32 -> f16 conversion (weights, done once)
// ---------------------------------------------------------------------------
__global__ __launch_bounds__(256) void cvt_f32_f16(_Float16* __restrict__ dst,
                                                   const float* __restrict__ src,
                                                   long long n) {
    long long i = (long long)blockIdx.x * blockDim.x + threadIdx.x;
    if (i < n) dst[i] = (_Float16)src[i];
}

// ---------------------------------------------------------------------------
// fc1: h = x @ fc1_w + fc1_b  -> f32 and f16 copies
// ---------------------------------------------------------------------------
__global__ __launch_bounds__(256) void fc1_kernel(const float* __restrict__ x,
                                                  const float* __restrict__ w,
                                                  const float* __restrict__ b,
                                                  float* __restrict__ h32,
                                                  _Float16* __restrict__ h16,
                                                  int N) {
    int idx = blockIdx.x * 256 + threadIdx.x;
    int n = idx >> 6, o = idx & 63;
    if (n >= N) return;
    float v = x[n] * w[o] + b[o];
    h32[idx] = v;
    h16[idx] = (_Float16)v;
}

// ---------------------------------------------------------------------------
// degree: deg[dst[e]] += 1
// ---------------------------------------------------------------------------
__global__ __launch_bounds__(256) void deg_kernel(const long long* __restrict__ dst,
                                                  float* __restrict__ deg, int E) {
    int e = blockIdx.x * 256 + threadIdx.x;
    if (e < E) atomicAdd(&deg[(int)dst[e]], 1.0f);
}

// ---------------------------------------------------------------------------
// kernel-MLP layer 1 (K=6, tiny): hk1 = relu(edge_attr @ k1_w + k1_b) -> f16
// ---------------------------------------------------------------------------
__global__ __launch_bounds__(256) void mlp1_kernel(const float* __restrict__ ea,
                                                   const float* __restrict__ w,
                                                   const float* __restrict__ b,
                                                   _Float16* __restrict__ out,
                                                   int E) {
    long long idx = (long long)blockIdx.x * 256 + threadIdx.x;
    int e = (int)(idx >> 10), c = (int)(idx & 1023);
    if (e >= E) return;
    float acc = b[c];
    const float* er = ea + (size_t)e * 6;
#pragma unroll
    for (int i = 0; i < 6; ++i) acc += er[i] * w[i * KW + c];
    out[idx] = (_Float16)fmaxf(acc, 0.0f);
}

// ---------------------------------------------------------------------------
// WMMA f16 GEMM:  Out[f16] = act( A[M,K](f16) @ B[K,Nn](f16) + bias )
//   block tile 128x128, 8 waves (4 M-strips x 2 N-strips), wave tile 32x64,
//   K stepped 32 (one v_wmma_f32_16x16x32_f16 per fragment pair per step).
//   Double-buffered LDS; A staged via async-to-LDS (ASYNCcnt), B staged via
//   software-pipelined register copies with transpose into Bs[n][k].
//   PERM: store with column permutation n=(d*64+o) -> row offset (o*64+d),
//   producing the per-edge weight tensor in [E, o, d] order.
// ---------------------------------------------------------------------------
template <bool RELU, bool PERM>
__global__ __launch_bounds__(256) void gemm_f16(const _Float16* __restrict__ A,
                                                const _Float16* __restrict__ B,
                                                const float* __restrict__ bias,
                                                _Float16* __restrict__ Out,
                                                int M, int Nn, int K) {
    __shared__ _Float16 As[2][128][40];   // 32 K-halfs + 8 pad (conflict-free)
    __shared__ _Float16 Bs[2][128][40];   // transposed: Bs[buf][n][k]

    const int tid  = threadIdx.x;
    const int lane = tid & 31;
    const int wave = tid >> 5;
    const int wm   = wave & 3;    // M strip: 32 rows
    const int wn   = wave >> 2;   // N strip: 64 cols
    const int lr   = lane & 15;
    const int hi   = lane >> 4;

    const long long rowBase = (long long)blockIdx.x * 128;
    const int       colBase = blockIdx.y * 128;

    // --- per-thread staging coordinates (invariant over K) ---
    const int       ar   = tid >> 1;            // A: row in tile
    const int       acs  = (tid & 1) * 16;      // A: 16-half segment
    const long long agr  = rowBase + ar;
    const bool      aval = (agr < M);
    const _Float16* Ag   = A + (size_t)agr * K + acs;

    const int       bkk = tid >> 3;             // B: k-row in tile
    const int       bns = (tid & 7) * 16;       // B: 16-half n segment
    const _Float16* Bg  = B + (size_t)bkk * Nn + colBase + bns;

    const int nsteps = K / 32;

    // ---------------- prologue: stage K-step 0 into buffer 0 ----------------
    {
#if HAVE_ASYNC_LDS
        if (aval) {
            async_cp16(&As[0][ar][acs],     Ag);
            async_cp16(&As[0][ar][acs + 8], Ag + 8);
        } else {
            float4 z = {0, 0, 0, 0};
            *(float4*)&As[0][ar][acs]     = z;
            *(float4*)&As[0][ar][acs + 8] = z;
        }
#else
        float4 v0 = {0, 0, 0, 0}, v1 = {0, 0, 0, 0};
        if (aval) {
            const float4* p = (const float4*)Ag;
            v0 = p[0];
            v1 = p[1];
        }
        *(float4*)&As[0][ar][acs]     = v0;
        *(float4*)&As[0][ar][acs + 8] = v1;
#endif
        const float4* bp = (const float4*)Bg;
        float4 b0 = bp[0], b1 = bp[1];
        _Float16 tmp[16];
        *(float4*)&tmp[0] = b0;
        *(float4*)&tmp[8] = b1;
#pragma unroll
        for (int j = 0; j < 16; ++j) Bs[0][bns + j][bkk] = tmp[j];
#if HAVE_ASYNC_LDS
        wait_async0();
#endif
    }
    __syncthreads();

    v8f acc[2][4];
#pragma unroll
    for (int i = 0; i < 2; ++i)
#pragma unroll
        for (int j = 0; j < 4; ++j) acc[i][j] = (v8f){};

    // ---------------- main K loop: one barrier per step ----------------
    for (int kt = 0; kt < nsteps; ++kt) {
        const int  cur       = kt & 1;
        const int  nxt       = cur ^ 1;
        const int  k1        = (kt + 1) * 32;
        const bool have_next = (kt + 1) < nsteps;

        float4 nb0, nb1;                         // next B tile (registers)
#if !HAVE_ASYNC_LDS
        float4 na0, na1;                         // next A tile (registers)
#endif
        if (have_next) {
#if HAVE_ASYNC_LDS
            if (aval) {                          // ASYNCcnt copy, lands during WMMA
                async_cp16(&As[nxt][ar][acs],     Ag + k1);
                async_cp16(&As[nxt][ar][acs + 8], Ag + k1 + 8);
            } else {
                float4 z = {0, 0, 0, 0};
                *(float4*)&As[nxt][ar][acs]     = z;
                *(float4*)&As[nxt][ar][acs + 8] = z;
            }
#else
            na0 = (float4){0, 0, 0, 0};
            na1 = (float4){0, 0, 0, 0};
            if (aval) {
                const float4* p = (const float4*)(Ag + k1);
                na0 = p[0];
                na1 = p[1];
            }
#endif
            const float4* bp = (const float4*)(Bg + (size_t)k1 * Nn);
            nb0 = bp[0];
            nb1 = bp[1];
            if (kt + 2 < nsteps)
                __builtin_prefetch(Bg + (size_t)(k1 + 32) * Nn, 0, 1);
        }

        // --- fragments per the CDNA5 16-bit WMMA VGPR layouts ---
        v16h a[2], b[4];
#pragma unroll
        for (int mi = 0; mi < 2; ++mi) {
            const _Float16* ap = &As[cur][wm * 32 + mi * 16 + lr][0];
#pragma unroll
            for (int j = 0; j < 8; ++j) {        // A: K = hi*8+j and hi*8+16+j
                a[mi][j]     = ap[hi * 8 + j];
                a[mi][j + 8] = ap[hi * 8 + 16 + j];
            }
        }
#pragma unroll
        for (int ni = 0; ni < 4; ++ni) {
            const _Float16* bp = &Bs[cur][wn * 64 + ni * 16 + lr][0];
#pragma unroll
            for (int j = 0; j < 8; ++j) {        // B: K = hi*8+j and hi*8+16+j
                b[ni][j]     = bp[hi * 8 + j];
                b[ni][j + 8] = bp[hi * 8 + 16 + j];
            }
        }
#pragma unroll
        for (int mi = 0; mi < 2; ++mi)
#pragma unroll
            for (int ni = 0; ni < 4; ++ni)
                acc[mi][ni] = __builtin_amdgcn_wmma_f32_16x16x32_f16(
                    false, a[mi], false, b[ni], (short)0, acc[mi][ni], false, false);

        if (have_next) {
            _Float16 tmp[16];
            *(float4*)&tmp[0] = nb0;
            *(float4*)&tmp[8] = nb1;
#pragma unroll
            for (int j = 0; j < 16; ++j) Bs[nxt][bns + j][bkk] = tmp[j];
#if !HAVE_ASYNC_LDS
            *(float4*)&As[nxt][ar][acs]     = na0;
            *(float4*)&As[nxt][ar][acs + 8] = na1;
#else
            wait_async0();                       // A(nxt) fully in LDS
#endif
        }
        __syncthreads();                         // publish nxt, release cur
    }

    // --- epilogue: bias, optional relu, f16 store (optionally permuted) ---
#pragma unroll
    for (int mi = 0; mi < 2; ++mi)
#pragma unroll
        for (int ni = 0; ni < 4; ++ni)
#pragma unroll
            for (int i = 0; i < 8; ++i) {
                long long gm = rowBase + wm * 32 + mi * 16 + hi * 8 + i;
                int       gn = colBase + wn * 64 + ni * 16 + lr;
                if (gm < M) {
                    float v = acc[mi][ni][i] + bias[gn];
                    if (RELU) v = fmaxf(v, 0.0f);
                    size_t idx;
                    if (PERM) {
                        int d = gn >> 6, o = gn & 63;
                        idx = (size_t)gm * Nn + (size_t)(o * 64 + d);
                    } else {
                        idx = (size_t)gm * Nn + gn;
                    }
                    Out[idx] = (_Float16)v;
                }
            }
}

// ---------------------------------------------------------------------------
// msg + scatter: agg[dst[e]] += h16[src[e]] @ W[e]    (W stored [E, o, d] f16)
//   4 edges / block, 64 threads per edge (one per output channel o);
//   each thread streams one contiguous 128B row of W.
// ---------------------------------------------------------------------------
__global__ __launch_bounds__(256) void msg_kernel(const _Float16* __restrict__ h16,
                                                  const _Float16* __restrict__ Wm,
                                                  const long long* __restrict__ src,
                                                  const long long* __restrict__ dst,
                                                  float* __restrict__ agg, int E) {
    __shared__ _Float16 hs[4][64];
    int sub = threadIdx.x >> 6;
    int o   = threadIdx.x & 63;
    long long e = (long long)blockIdx.x * 4 + sub;
    if (e < E) {
        int s = (int)src[e];
        hs[sub][o] = h16[(size_t)s * WIDTH + o];
    }
    __syncthreads();
    if (e >= E) return;
    const _Float16* w = Wm + (size_t)e * (WIDTH * WIDTH) + (size_t)o * WIDTH;
    float acc = 0.0f;
#pragma unroll
    for (int d = 0; d < WIDTH; ++d) acc += (float)hs[sub][d] * (float)w[d];
    atomicAdd(&agg[(size_t)((int)dst[e]) * WIDTH + o], acc);
}

// ---------------------------------------------------------------------------
// node update: h = relu(agg/denom + h @ root + conv_b);  root staged in LDS
// ---------------------------------------------------------------------------
__global__ __launch_bounds__(256) void update_kernel(const float* __restrict__ hIn,
                                                     const float* __restrict__ agg,
                                                     const float* __restrict__ deg,
                                                     const float* __restrict__ root,
                                                     const float* __restrict__ cb,
                                                     float* __restrict__ hOut,
                                                     _Float16* __restrict__ h16,
                                                     int N) {
    __shared__ float rootS[64][64];
    __shared__ float hs[4][64];
    int t = threadIdx.x;
#pragma unroll
    for (int i = t; i < 4096; i += 256) rootS[i >> 6][i & 63] = root[i];
    int sub  = t >> 6;
    int o    = t & 63;
    int node = blockIdx.x * 4 + sub;
    if (node < N) hs[sub][o] = hIn[(size_t)node * WIDTH + o];
    __syncthreads();
    if (node >= N) return;
    float dg = deg[node];
    float dn = dg > 1.0f ? dg : 1.0f;
    float acc = cb[o] + agg[(size_t)node * WIDTH + o] / dn;
    const float* hr = hs[sub];
#pragma unroll
    for (int d = 0; d < WIDTH; ++d) acc += hr[d] * rootS[d][o];
    acc = fmaxf(acc, 0.0f);
    hOut[(size_t)node * WIDTH + o] = acc;
    h16[(size_t)node * WIDTH + o]  = (_Float16)acc;
}

// ---------------------------------------------------------------------------
// fc2: out = h @ fc2_w + fc2_b
// ---------------------------------------------------------------------------
__global__ __launch_bounds__(256) void fc2_kernel(const float* __restrict__ h,
                                                  const float* __restrict__ w,
                                                  const float* __restrict__ b,
                                                  float* __restrict__ out, int N) {
    int n = blockIdx.x * 256 + threadIdx.x;
    if (n >= N) return;
    float acc = b[0];
    const float* hr = h + (size_t)n * WIDTH;
#pragma unroll
    for (int o = 0; o < WIDTH; ++o) acc += hr[o] * w[o];
    out[n] = acc;
}

// ---------------------------------------------------------------------------
extern "C" void kernel_launch(void* const* d_in, const int* in_sizes, int n_in,
                              void* d_out, int out_size, void* d_ws, size_t ws_size,
                              hipStream_t stream) {
    const float*     x      = (const float*)d_in[0];
    const long long* eidx   = (const long long*)d_in[1];   // jnp.int64
    const float*     eattr  = (const float*)d_in[2];
    const float*     fc1_w  = (const float*)d_in[3];
    const float*     fc1_b  = (const float*)d_in[4];
    const float*     k1_w   = (const float*)d_in[5];
    const float*     k1_b   = (const float*)d_in[6];
    const float*     k2_w   = (const float*)d_in[7];
    const float*     k2_b   = (const float*)d_in[8];
    const float*     k3_w   = (const float*)d_in[9];
    const float*     k3_b   = (const float*)d_in[10];
    const float*     root   = (const float*)d_in[11];
    const float*     conv_b = (const float*)d_in[12];
    const float*     fc2_w  = (const float*)d_in[13];
    const float*     fc2_b  = (const float*)d_in[14];

    const int N = in_sizes[0];
    const int E = in_sizes[1] / 2;
    const long long* src = eidx;
    const long long* dst = eidx + E;

    // ---- carve workspace ----
    char*  ws  = (char*)d_ws;
    size_t off = 0;
    auto carve = [&](size_t bytes) -> char* {
        off = (off + 255) & ~(size_t)255;
        char* p = ws + off;
        off += bytes;
        return p;
    };
    _Float16* hk1 = (_Float16*)carve((size_t)E * KW * sizeof(_Float16));
    _Float16* hk2 = (_Float16*)carve((size_t)E * KW * sizeof(_Float16));
    _Float16* Wm  = (_Float16*)carve((size_t)E * WIDTH * WIDTH * sizeof(_Float16));
    _Float16* k2h = (_Float16*)carve((size_t)KW * KW * sizeof(_Float16));
    _Float16* k3h = (_Float16*)carve((size_t)KW * WIDTH * WIDTH * sizeof(_Float16));
    float*    hA  = (float*)carve((size_t)N * WIDTH * sizeof(float));
    float*    hB  = (float*)carve((size_t)N * WIDTH * sizeof(float));
    _Float16* h16 = (_Float16*)carve((size_t)N * WIDTH * sizeof(_Float16));
    float*    agg = (float*)carve((size_t)N * WIDTH * sizeof(float));
    float*    deg = (float*)carve((size_t)N * sizeof(float));
    (void)ws_size;

    // ---- one-time precompute ----
    cvt_f32_f16<<<(KW * KW + 255) / 256, 256, 0, stream>>>(k2h, k2_w, (long long)KW * KW);
    cvt_f32_f16<<<(KW * WIDTH * WIDTH + 255) / 256, 256, 0, stream>>>(
        k3h, k3_w, (long long)KW * WIDTH * WIDTH);

    fc1_kernel<<<((size_t)N * WIDTH + 255) / 256, 256, 0, stream>>>(x, fc1_w, fc1_b, hA, h16, N);

    (void)hipMemsetAsync(deg, 0, (size_t)N * sizeof(float), stream);
    deg_kernel<<<(E + 255) / 256, 256, 0, stream>>>(dst, deg, E);

    mlp1_kernel<<<(unsigned)(((size_t)E * KW + 255) / 256), 256, 0, stream>>>(
        eattr, k1_w, k1_b, hk1, E);

    // ---- the two big WMMA GEMMs ----
    {
        dim3 g((E + 127) / 128, KW / 128);
        gemm_f16<true, false><<<g, 256, 0, stream>>>(hk1, k2h, k2_b, hk2, E, KW, KW);
    }
    {
        dim3 g((E + 127) / 128, (WIDTH * WIDTH) / 128);
        gemm_f16<false, true><<<g, 256, 0, stream>>>(hk2, k3h, k3_b, Wm, E, WIDTH * WIDTH, KW);
    }

    // ---- 6 message-passing iterations ----
    float* hin  = hA;
    float* hout = hB;
    for (int it = 0; it < 6; ++it) {
        (void)hipMemsetAsync(agg, 0, (size_t)N * WIDTH * sizeof(float), stream);
        msg_kernel<<<(E + 3) / 4, 256, 0, stream>>>(h16, Wm, src, dst, agg, E);
        update_kernel<<<(N + 3) / 4, 256, 0, stream>>>(hin, agg, deg, root, conv_b, hout, h16, N);
        float* t = hin; hin = hout; hout = t;
    }

    // ---- readout ----
    fc2_kernel<<<(N + 255) / 256, 256, 0, stream>>>(hin, fc2_w, fc2_b, (float*)d_out, N);
}
